// Attention_36739150250506
// MI455X (gfx1250) — matrix operand order repaired and checked by reading, not compiled, not run on hardware
//
#include <hip/hip_runtime.h>

// ---------------------------------------------------------------------------
// MI455X (gfx1250) attention layer: bf16 WMMA + branch-free 3-stage async
// global->LDS pipelines (unconditional s_wait_asynccnt 4, clamped prefetch).
// S=2048, H=2048, NH=16, NKV=4, D=128 (hardcoded to the reference shapes).
// ---------------------------------------------------------------------------

typedef __attribute__((ext_vector_type(16))) __bf16 bf16x16;
typedef __attribute__((ext_vector_type(8)))  float  f32x8;

#define S_LEN 2048
#define H_DIM 2048
#define NH    16
#define NKV   4
#define D_HD  128

__device__ __forceinline__ unsigned short f2bfu(float f) {
    union { float f; unsigned u; } c; c.f = f;
    unsigned u = c.u;
    return (unsigned short)((u + 0x7FFFu + ((u >> 16) & 1u)) >> 16);
}

__device__ __forceinline__ f32x8 zero8() {
    f32x8 z;
#pragma unroll
    for (int i = 0; i < 8; ++i) z[i] = 0.0f;
    return z;
}

// CDNA5 async copy: 16B per lane, global -> LDS, tracked by ASYNCcnt.
__device__ __forceinline__ void async_copy_b128(const unsigned short* g,
                                                unsigned short* l) {
    unsigned lds = (unsigned)(uintptr_t)l;               // low 32b = LDS offset
    unsigned long long ga = (unsigned long long)(uintptr_t)g;
    asm volatile("global_load_async_to_lds_b128 %0, %1, off"
                 :: "v"(lds), "v"(ga) : "memory");
}
// Retire the oldest 4 copies (in-order completion); keep newest 4 in flight.
__device__ __forceinline__ void wait_async4() {
    asm volatile("s_wait_asynccnt 4" ::: "memory");
}

// A-fragment (16x32 bf16, MxK). Lanes 0-15 hold K=0..7,16..23;
// lanes 16-31 hold K=8..15,24..31 for row m = lane%16.
__device__ __forceinline__ bf16x16 ld_frag_a(const unsigned short* p, int stride, int k0) {
    const int lane = threadIdx.x & 31;
    const int row  = lane & 15;
    const int hi   = (lane >> 4) & 1;
    union { bf16x16 v; uint4 q[2]; } u;
    u.q[0] = *(const uint4*)(p + row * stride + k0 + hi * 8);
    u.q[1] = *(const uint4*)(p + row * stride + k0 + 16 + hi * 8);
    return u.v;
}

// B-fragment (32x16 bf16, KxN), source stored as [n][k] rows (NT layout).
__device__ __forceinline__ bf16x16 ld_frag_b(const unsigned short* p, int stride, int k0) {
    const int lane = threadIdx.x & 31;
    const int col  = lane & 15;
    const int hi   = (lane >> 4) & 1;
    const unsigned short* q = p + col * stride + k0 + hi * 8;
    union { bf16x16 v; uint4 q[2]; } u;
    u.q[0] = *(const uint4*)(q);
    u.q[1] = *(const uint4*)(q + 16);
    return u.v;
}

__device__ __forceinline__ f32x8 wmma_bf16(bf16x16 a, bf16x16 b, f32x8 c) {
    return __builtin_amdgcn_wmma_f32_16x16x32_bf16(false, a, false, b, (short)0, c, false, false);
}

// ---------------------------------------------------------------------------
// fp32 -> bf16 conversion
// ---------------------------------------------------------------------------
__global__ void cvt_bf16_kernel(const float* __restrict__ in,
                                unsigned short* __restrict__ out, int n) {
    int i = blockIdx.x * blockDim.x + threadIdx.x;
    if (i < n) out[i] = f2bfu(in[i]);
}

// V: [s][h*128+d] f32 -> transposed [h][d][s] bf16 (PV B-fragments become
// contiguous so async tile loads need no in-LDS transpose).
__global__ void vtcvt_kernel(const float* __restrict__ vf,
                             unsigned short* __restrict__ vt) {
    int idx = blockIdx.x * blockDim.x + threadIdx.x;  // (h*128+d)*2048 + s
    if (idx >= NKV * D_HD * S_LEN) return;
    int s = idx & (S_LEN - 1);
    int d = (idx >> 11) & 127;
    int h = idx >> 18;
    vt[idx] = f2bfu(vf[(size_t)s * (NKV * D_HD) + h * D_HD + d]);
}

// ---------------------------------------------------------------------------
// NT GEMM: C[M,N] f32 = A[M,K] * B[N,K]^T, bf16 operands.
// Block tile 128x128, 8 waves (wave tile 32x64). K staged 32/iter through
// triple-buffered LDS; branch-free steady state: prefetch for tile it+2 is
// issued every iteration (clamped at the end), so s_wait_asynccnt 4 always
// retires exactly tile `it` while tile it+1 stays in flight.
// ---------------------------------------------------------------------------
__global__ __launch_bounds__(256) void gemm_nt_bf16(
    const unsigned short* __restrict__ A, const unsigned short* __restrict__ B,
    float* __restrict__ C, int M, int N, int K) {
    __shared__ __align__(16) unsigned short At[3][128 * 32];
    __shared__ __align__(16) unsigned short Bt[3][128 * 32];

    const int tid  = threadIdx.x;
    const int lane = tid & 31;
    const int wave = tid >> 5;
    const int hi   = (lane >> 4) & 1;
    const int bm   = blockIdx.y;
    const int bn   = blockIdx.x;
    const int wrow = (wave & 3) * 32;   // 0,32,64,96
    const int wcol = (wave >> 2) * 64;  // 0,64

    auto issue_tiles = [&](int k0, int buf) {  // 4 async copies per thread
#pragma unroll
        for (int p = 0; p < 2; ++p) {
            int idx = tid + p * 256;          // 0..511
            int row = idx >> 2, c = idx & 3;  // 128 rows x 4 chunks
            async_copy_b128(A + (size_t)(bm * 128 + row) * K + k0 + c * 8,
                            &At[buf][row * 32 + c * 8]);
            async_copy_b128(B + (size_t)(bn * 128 + row) * K + k0 + c * 8,
                            &Bt[buf][row * 32 + c * 8]);
        }
    };

    f32x8 acc[2][4];
#pragma unroll
    for (int mi = 0; mi < 2; ++mi)
#pragma unroll
        for (int ni = 0; ni < 4; ++ni) acc[mi][ni] = zero8();

    const int niter = K >> 5;   // >= 2 for all our calls (K = 2048)
    issue_tiles(0, 0);
    issue_tiles(32, 1);
    int cur = 0, nxt = 2;
    for (int it = 0; it < niter; ++it) {
        wait_async4();          // tile `it` resident; tile it+1 in flight
        __syncthreads();
        int k2 = it + 2;
        if (k2 >= niter) k2 = niter - 1;   // clamped dummy prefetch at tail
        issue_tiles(k2 << 5, nxt);

        bf16x16 a0 = ld_frag_a(&At[cur][(wrow + 0) * 32], 32, 0);
        bf16x16 a1 = ld_frag_a(&At[cur][(wrow + 16) * 32], 32, 0);
#pragma unroll
        for (int ni = 0; ni < 4; ++ni) {
            bf16x16 b = ld_frag_b(&Bt[cur][(wcol + ni * 16) * 32], 32, 0);
            acc[0][ni] = wmma_bf16(a0, b, acc[0][ni]);
            acc[1][ni] = wmma_bf16(a1, b, acc[1][ni]);
        }
        cur = (cur == 2) ? 0 : cur + 1;
        nxt = (nxt == 2) ? 0 : nxt + 1;
    }

#pragma unroll
    for (int mi = 0; mi < 2; ++mi)
#pragma unroll
        for (int ni = 0; ni < 4; ++ni)
#pragma unroll
            for (int r = 0; r < 8; ++r) {
                int gm = bm * 128 + wrow + mi * 16 + r + 8 * hi;
                int gn = bn * 128 + wcol + ni * 16 + (lane & 15);
                C[(size_t)gm * N + gn] = acc[mi][ni][r];
            }
}

// ---------------------------------------------------------------------------
// Fused RMSNorm(D=128) + RoPE per (s, head) row.
// in: [s][nh*128 + d] f32 ; out: [h][s][d] bf16
// ---------------------------------------------------------------------------
__global__ __launch_bounds__(128) void norm_rope_kernel(
    const float* __restrict__ X, const float* __restrict__ sinp,
    const float* __restrict__ cosp, const float* __restrict__ w,
    unsigned short* __restrict__ out, int nh) {
    __shared__ float sh[128];
    __shared__ float red[128];
    const int s = blockIdx.x, h = blockIdx.y, d = threadIdx.x;

    float x = X[(size_t)s * (nh * D_HD) + h * D_HD + d];
    red[d] = x * x;
    __syncthreads();
#pragma unroll
    for (int off = 64; off > 0; off >>= 1) {
        if (d < off) red[d] += red[d + off];
        __syncthreads();
    }
    float inv = rsqrtf(red[0] * (1.0f / 128.0f) + 1.1920928955078125e-07f);
    float xn = x * inv * w[d];
    sh[d] = xn;
    __syncthreads();
    float rot = (d < 64) ? -sh[d + 64] : sh[d - 64];
    float y = cosp[s * D_HD + d] * xn + sinp[s * D_HD + d] * rot;
    out[((size_t)h * S_LEN + s) * D_HD + d] = f2bfu(y);
}

// ---------------------------------------------------------------------------
// Flash attention: block = (head, 128 q rows), 8 waves x 16 q rows.
// KV tiles of 32 triple-buffered via async copies (K as [kv][d], V
// pre-transposed in global to [d][s] per head). Branch-free pipeline as in
// the GEMM. Online softmax in fp32; P routed through per-wave LDS scratch.
// ---------------------------------------------------------------------------
__global__ __launch_bounds__(256) void flash_attn_kernel(
    const unsigned short* __restrict__ qh,  // [NH][S][D]
    const unsigned short* __restrict__ kh,  // [NKV][S][D]
    const unsigned short* __restrict__ vt,  // [NKV][D][S]
    unsigned short* __restrict__ ob)        // [S][H]
{
    __shared__ __align__(16) unsigned short Kt[3][32 * 128];  // [kv][d]
    __shared__ __align__(16) unsigned short Vt[3][128 * 32];  // [d][kv]
    __shared__ __align__(16) unsigned short Pb[8 * 16 * 32];

    const int tid  = threadIdx.x;
    const int lane = tid & 31;
    const int wave = tid >> 5;
    const int hi   = (lane >> 4) & 1;
    const int h    = blockIdx.y;
    const int hk   = h >> 2;               // GQA repeat_interleave(4)
    const int qb   = blockIdx.x * 128;
    const int qr   = qb + wave * 16;
    const float scale = 0.08838834764831845f;  // 1/sqrt(128)

    auto issue_kv = [&](int kb, int buf) {  // 4 async copies per thread
#pragma unroll
        for (int p = 0; p < 2; ++p) {
            int idx = tid + p * 256;            // 0..511
            int kr = idx >> 4, kc = idx & 15;   // 32 rows x 16 chunks of K
            async_copy_b128(kh + ((size_t)hk * S_LEN + kb + kr) * D_HD + kc * 8,
                            &Kt[buf][kr * 128 + kc * 8]);
            int d = idx >> 2, vc = idx & 3;     // 128 rows x 4 chunks of V^T
            async_copy_b128(vt + ((size_t)hk * D_HD + d) * S_LEN + kb + vc * 8,
                            &Vt[buf][d * 32 + vc * 8]);
        }
    };

    const unsigned short* qbase = qh + ((size_t)h * S_LEN + qr) * D_HD;
    bf16x16 qf[4];
#pragma unroll
    for (int ks = 0; ks < 4; ++ks) qf[ks] = ld_frag_a(qbase, D_HD, ks * 32);

    f32x8 oacc[8];
#pragma unroll
    for (int j = 0; j < 8; ++j) oacc[j] = zero8();
    float m_i[8], l_i[8];
#pragma unroll
    for (int r = 0; r < 8; ++r) { m_i[r] = -3.0e38f; l_i[r] = 0.0f; }

    unsigned short* pw = &Pb[wave * 16 * 32];

    const int niter = (qb + 128) >> 5;   // always >= 4
    issue_kv(0, 0);
    issue_kv(32, 1);
    int cur = 0, nxt = 2;
    for (int it = 0; it < niter; ++it) {
        const int kb = it << 5;
        wait_async4();
        __syncthreads();
        int k2 = it + 2;
        if (k2 >= niter) k2 = niter - 1;   // clamped dummy prefetch at tail
        issue_kv(k2 << 5, nxt);

        if (kb <= qr + 15) {
            // ---- S = Q K^T (16 x 32 per wave) ----
            f32x8 sfr[2];
#pragma unroll
            for (int t = 0; t < 2; ++t) {
                f32x8 z = zero8();
#pragma unroll
                for (int ks = 0; ks < 4; ++ks) {
                    bf16x16 b = ld_frag_b(&Kt[cur][(t * 16) * 128], 128, ks * 32);
                    z = wmma_bf16(qf[ks], b, z);
                }
#pragma unroll
                for (int r = 0; r < 8; ++r) z[r] *= scale;
                sfr[t] = z;
            }
            // causal mask (tiles straddling the diagonal)
            if (kb + 31 > qr) {
#pragma unroll
                for (int t = 0; t < 2; ++t)
#pragma unroll
                    for (int r = 0; r < 8; ++r) {
                        int kcol = kb + t * 16 + (lane & 15);
                        int row  = qr + r + 8 * hi;
                        if (kcol > row) sfr[t][r] = -3.0e38f;
                    }
            }
            // ---- online softmax ----
#pragma unroll
            for (int r = 0; r < 8; ++r) {
                float mx = fmaxf(sfr[0][r], sfr[1][r]);
#pragma unroll
                for (int off = 8; off > 0; off >>= 1)
                    mx = fmaxf(mx, __shfl_xor(mx, off));
                float mnew = fmaxf(m_i[r], mx);
                float corr = __expf(m_i[r] - mnew);
                float rs = 0.0f;
#pragma unroll
                for (int t = 0; t < 2; ++t) {
                    float pv = __expf(sfr[t][r] - mnew);
                    sfr[t][r] = pv;
                    rs += pv;
                }
#pragma unroll
                for (int off = 8; off > 0; off >>= 1)
                    rs += __shfl_xor(rs, off);
                l_i[r] = l_i[r] * corr + rs;
                m_i[r] = mnew;
#pragma unroll
                for (int j = 0; j < 8; ++j) oacc[j][r] *= corr;
            }
            // ---- spill P and rebuild A-fragment ----
#pragma unroll
            for (int t = 0; t < 2; ++t)
#pragma unroll
                for (int r = 0; r < 8; ++r)
                    pw[(r + 8 * hi) * 32 + t * 16 + (lane & 15)] = f2bfu(sfr[t][r]);
            bf16x16 pa = ld_frag_a(pw, 32, 0);
#pragma unroll
            for (int j = 0; j < 8; ++j) {
                bf16x16 vb = ld_frag_b(&Vt[cur][(j * 16) * 32], 32, 0);
                oacc[j] = wmma_bf16(pa, vb, oacc[j]);
            }
        }
        cur = (cur == 2) ? 0 : cur + 1;
        nxt = (nxt == 2) ? 0 : nxt + 1;
    }

#pragma unroll
    for (int r = 0; r < 8; ++r) {
        float inv = 1.0f / l_i[r];
        int s = qr + r + 8 * hi;
#pragma unroll
        for (int j = 0; j < 8; ++j) {
            int col = h * D_HD + j * 16 + (lane & 15);
            ob[(size_t)s * H_DIM + col] = f2bfu(oacc[j][r] * inv);
        }
    }
}

// ---------------------------------------------------------------------------
// Orchestration
// ---------------------------------------------------------------------------
extern "C" void kernel_launch(void* const* d_in, const int* in_sizes, int n_in,
                              void* d_out, int out_size, void* d_ws, size_t ws_size,
                              hipStream_t stream) {
    const float* x    = (const float*)d_in[0];
    const float* sinp = (const float*)d_in[1];
    const float* cosp = (const float*)d_in[2];
    // d_in[3] = mask (causal; computed analytically in-kernel)
    const float* wq   = (const float*)d_in[4];
    const float* wk   = (const float*)d_in[5];
    const float* wv   = (const float*)d_in[6];
    const float* wo   = (const float*)d_in[7];
    const float* qnw  = (const float*)d_in[8];
    const float* knw  = (const float*)d_in[9];
    float* out = (float*)d_out;

    char* ws = (char*)d_ws;
    size_t off = 0;
    auto alloc = [&](size_t bytes) -> void* {
        void* p = ws + off;
        off += (bytes + 255) & ~(size_t)255;
        return p;
    };
    const size_t E4 = (size_t)S_LEN * H_DIM;      // 4M elems
    const size_t E1 = (size_t)S_LEN * NKV * D_HD; // 1M elems

    unsigned short* xb  = (unsigned short*)alloc(E4 * 2);
    unsigned short* wqb = (unsigned short*)alloc(E4 * 2);
    unsigned short* wkb = (unsigned short*)alloc(E1 * 2);
    unsigned short* wvb = (unsigned short*)alloc(E1 * 2);
    unsigned short* wob = (unsigned short*)alloc(E4 * 2);
    float*          Qf  = (float*)alloc(E4 * 4);
    float*          Kf  = (float*)alloc(E1 * 4);
    float*          Vf  = (float*)alloc(E1 * 4);
    unsigned short* qhB = (unsigned short*)alloc(E4 * 2);
    unsigned short* khB = (unsigned short*)alloc(E1 * 2);
    unsigned short* vtB = (unsigned short*)alloc(E1 * 2);
    unsigned short* obB = (unsigned short*)alloc(E4 * 2);

    // 1) fp32 -> bf16
    cvt_bf16_kernel<<<(E4 + 255) / 256, 256, 0, stream>>>(x, xb, (int)E4);
    cvt_bf16_kernel<<<(E4 + 255) / 256, 256, 0, stream>>>(wq, wqb, (int)E4);
    cvt_bf16_kernel<<<(E1 + 255) / 256, 256, 0, stream>>>(wk, wkb, (int)E1);
    cvt_bf16_kernel<<<(E1 + 255) / 256, 256, 0, stream>>>(wv, wvb, (int)E1);
    cvt_bf16_kernel<<<(E4 + 255) / 256, 256, 0, stream>>>(wo, wob, (int)E4);

    // 2) projections (NT GEMMs)
    gemm_nt_bf16<<<dim3(H_DIM / 128, S_LEN / 128), 256, 0, stream>>>(xb, wqb, Qf, S_LEN, H_DIM, H_DIM);
    gemm_nt_bf16<<<dim3((NKV * D_HD) / 128, S_LEN / 128), 256, 0, stream>>>(xb, wkb, Kf, S_LEN, NKV * D_HD, H_DIM);
    gemm_nt_bf16<<<dim3((NKV * D_HD) / 128, S_LEN / 128), 256, 0, stream>>>(xb, wvb, Vf, S_LEN, NKV * D_HD, H_DIM);

    // 3) RMSNorm + RoPE -> bf16 head-major layouts; V transposed per head
    norm_rope_kernel<<<dim3(S_LEN, NH), 128, 0, stream>>>(Qf, sinp, cosp, qnw, qhB, NH);
    norm_rope_kernel<<<dim3(S_LEN, NKV), 128, 0, stream>>>(Kf, sinp, cosp, knw, khB, NKV);
    vtcvt_kernel<<<(E1 + 255) / 256, 256, 0, stream>>>(Vf, vtB);

    // 4) causal flash attention
    flash_attn_kernel<<<dim3(S_LEN / 128, NH), 256, 0, stream>>>(qhB, khB, vtB, obB);

    // 5) output projection -> d_out (fp32)
    gemm_nt_bf16<<<dim3(H_DIM / 128, S_LEN / 128), 256, 0, stream>>>(obB, wob, out, S_LEN, H_DIM, H_DIM);
}
